// VOXCPM_FEAT_ENCODER_47296179864181
// MI455X (gfx1250) — compile-verified, matrix operand order checked by reading, not run on hardware
//
#include <hip/hip_runtime.h>
#include <hip/hip_bf16.h>

// ---------------- problem constants ----------------
#define LAY   4
#define HDIM  1024
#define NHEAD 16
#define KVH   2
#define HD    64
#define FF    4096
#define PATCH 4
#define QLEN  5
#define NSEQ  8192
#define DLM   2048
#define NTOK  (NSEQ * QLEN)       // 40960
#define MROWS (NSEQ * PATCH)      // 32768

typedef __attribute__((ext_vector_type(16))) __bf16 bf16x16;
typedef __attribute__((ext_vector_type(8)))  float  f32x8;

__device__ __forceinline__ unsigned short f2bf(float x) {
  unsigned u = __builtin_bit_cast(unsigned, x);
  u += 0x7fffu + ((u >> 16) & 1u);          // round-to-nearest-even
  return (unsigned short)(u >> 16);
}
__device__ __forceinline__ float bf2f(unsigned short h) {
  unsigned u = ((unsigned)h) << 16;
  return __builtin_bit_cast(float, u);
}

// gfx1250 async global->LDS copy, 16 bytes per lane (tracked by ASYNCcnt)
__device__ __forceinline__ void async_copy16(unsigned lds_addr, const unsigned short* gaddr) {
  asm volatile("global_load_async_to_lds_b128 %0, %1, off"
               :: "v"(lds_addr), "v"(gaddr) : "memory");
}

// ---------------- WMMA GEMM ----------------
// C[M,N] = A[M,K](bf16, row-major) * B[N,K](bf16, weight pre-transposed)
// Block tile 128(M) x 256(N), 8 waves as 2x4; wave tile 64x64 = 4x4 WMMA tiles.
// K-step 32; double-buffered LDS filled with global_load_async_to_lds_b128.
#define BTM  128
#define BTN  256
#define KT   32
#define LDST 40   // lds row stride (elements); 80B rows -> conflict-free b128 reads

enum { EPI_NONE = 0, EPI_RES = 1, EPI_SILU = 2, EPI_MUL = 3 };

template <int EPI>
__global__ __launch_bounds__(256)
void gemm_wmma(const unsigned short* __restrict__ A,
               const unsigned short* __restrict__ B,
               float* __restrict__ Cf,            // EPI_NONE / EPI_RES
               const float* __restrict__ Res,     // EPI_RES
               unsigned short* __restrict__ Cbf,  // EPI_SILU / EPI_MUL (in-place for MUL)
               int M, int N, int K)
{
  __shared__ __align__(16) unsigned short ldsA[2][BTM * LDST];
  __shared__ __align__(16) unsigned short ldsB[2][BTN * LDST];

  const int tid  = threadIdx.x;
  const int bm0  = blockIdx.y * BTM;
  const int bn0  = blockIdx.x * BTN;
  const int wave = tid >> 5;
  const int lane = tid & 31;
  const int half = lane >> 4;      // K-half select per ISA 16-bit fragment layout
  const int r    = lane & 15;      // row (A) / col (B) within 16x16 tile
  const int wm0  = (wave & 1) * 64;
  const int wn0  = (wave >> 1) * 64;

  f32x8 acc[4][4];
#pragma unroll
  for (int i = 0; i < 4; ++i)
#pragma unroll
    for (int j = 0; j < 4; ++j)
#pragma unroll
      for (int e = 0; e < 8; ++e) acc[i][j][e] = 0.0f;

  // issue one K-tile of async copies: A 512 chunks, B 1024 chunks, 16B each
  auto issue_tile = [&](int kk, int buf) {
#pragma unroll
    for (int j = 0; j < 2; ++j) {
      int chunk = tid + j * 256;          // 0..511
      int row   = chunk >> 2;             // 4 chunks per 32-elem row
      int col   = (chunk & 3) * 8;
      unsigned la = (unsigned)(size_t)&ldsA[buf][row * LDST + col];
      async_copy16(la, A + (size_t)(bm0 + row) * K + kk + col);
    }
#pragma unroll
    for (int j = 0; j < 4; ++j) {
      int chunk = tid + j * 256;          // 0..1023
      int row   = chunk >> 2;
      int col   = (chunk & 3) * 8;
      unsigned lb = (unsigned)(size_t)&ldsB[buf][row * LDST + col];
      async_copy16(lb, B + (size_t)(bn0 + row) * K + kk + col);
    }
  };

  const int nk = K / KT;
  issue_tile(0, 0);

  for (int kt = 0; kt < nk; ++kt) {
    const int cur = kt & 1;
    const bool hasNext = (kt + 1 < nk);

    __syncthreads();                       // next buffer free (prev compute done)
    if (hasNext) {
      issue_tile((kt + 1) * KT, cur ^ 1);
      asm volatile("s_wait_asynccnt 0x6" ::: "memory");  // current tile landed
    } else {
      asm volatile("s_wait_asynccnt 0x0" ::: "memory");
    }
    __syncthreads();                       // current tile visible to all waves

    union FragU { bf16x16 v; uint4 q[2]; } afr[4], bfr[4];
    // 16-bit fragment layout (ISA 7.12.2): lane r (+half) holds K = half*8+j
    // (j<8) and K = 16+half*8+(j-8) -> two contiguous 16B chunks.
#pragma unroll
    for (int mt = 0; mt < 4; ++mt) {
      const unsigned short* p = &ldsA[cur][(wm0 + mt * 16 + r) * LDST];
      afr[mt].q[0] = *(const uint4*)(p + half * 8);
      afr[mt].q[1] = *(const uint4*)(p + 16 + half * 8);
    }
#pragma unroll
    for (int nt = 0; nt < 4; ++nt) {
      const unsigned short* p = &ldsB[cur][(wn0 + nt * 16 + r) * LDST];
      bfr[nt].q[0] = *(const uint4*)(p + half * 8);
      bfr[nt].q[1] = *(const uint4*)(p + 16 + half * 8);
    }
#pragma unroll
    for (int mt = 0; mt < 4; ++mt)
#pragma unroll
      for (int nt = 0; nt < 4; ++nt)
        acc[mt][nt] = __builtin_amdgcn_wmma_f32_16x16x32_bf16(
            false, afr[mt].v, false, bfr[nt].v, (short)0, acc[mt][nt], false, false);
  }

  // epilogue: C/D layout (ISA): lane half h, elem i -> row = 16*mt + 8*h + i, col = r
#pragma unroll
  for (int mt = 0; mt < 4; ++mt) {
#pragma unroll
    for (int nt = 0; nt < 4; ++nt) {
      int col = bn0 + wn0 + nt * 16 + r;
#pragma unroll
      for (int i = 0; i < 8; ++i) {
        int row    = bm0 + wm0 + mt * 16 + half * 8 + i;
        size_t idx = (size_t)row * N + col;
        float v    = acc[mt][nt][i];
        if (EPI == EPI_NONE) {
          Cf[idx] = v;
        } else if (EPI == EPI_RES) {
          Cf[idx] = Res[idx] + v;
        } else if (EPI == EPI_SILU) {
          Cbf[idx] = f2bf(v / (1.0f + __expf(-v)));
        } else {  // EPI_MUL: act = silu(gate) * up, in place
          float g  = bf2f(Cbf[idx]);
          Cbf[idx] = f2bf(g * v);
        }
      }
    }
  }
}

// ---------------- RMSNorm -> bf16 ----------------
__global__ __launch_bounds__(256)
void rms_bf16(const float* __restrict__ in, int rowStride,
              const float* __restrict__ w, unsigned short* __restrict__ out)
{
  __shared__ float red[256];
  const int row = blockIdx.x;
  const float* src = in + (size_t)row * rowStride;
  float v[4];
  float ss = 0.0f;
#pragma unroll
  for (int i = 0; i < 4; ++i) {
    v[i] = src[threadIdx.x + i * 256];
    ss += v[i] * v[i];
  }
  red[threadIdx.x] = ss;
  __syncthreads();
  for (int s = 128; s > 0; s >>= 1) {
    if (threadIdx.x < s) red[threadIdx.x] += red[threadIdx.x + s];
    __syncthreads();
  }
  float inv = rsqrtf(red[0] * (1.0f / 1024.0f) + 1e-5f);
#pragma unroll
  for (int i = 0; i < 4; ++i) {
    int c = threadIdx.x + i * 256;
    out[(size_t)row * HDIM + c] = f2bf(v[i] * inv * w[c]);
  }
}

// ---------------- weight convert / transpose ----------------
// src f32 [K][Ncols] row-major  ->  dst bf16 [Ncols][K]
__global__ void tconv(const float* __restrict__ src, unsigned short* __restrict__ dst,
                      int K, int Ncols, int total)
{
  int t = blockIdx.x * 256 + threadIdx.x;
  if (t >= total) return;
  int k = t / Ncols, n = t % Ncols;
  dst[(size_t)n * K + k] = f2bf(src[t]);
}

__global__ void conv_bf16(const float* __restrict__ src, unsigned short* __restrict__ dst,
                          int total)
{
  int t = blockIdx.x * 256 + threadIdx.x;
  if (t < total) dst[t] = f2bf(src[t]);
}

// ---------------- assemble h = [special ; audio @ in_proj] ----------------
__global__ void build_h(const float* __restrict__ tmp,  // [NSEQ*PATCH, HDIM]
                        const float* __restrict__ sp,   // [HDIM]
                        float* __restrict__ h)          // [NTOK, HDIM]
{
  int t    = blockIdx.x * 256 + threadIdx.x;  // < NTOK*HDIM
  int c    = t & (HDIM - 1);
  int rowq = t >> 10;
  int seq  = rowq / QLEN;
  int tok  = rowq % QLEN;
  h[t] = (tok == 0) ? sp[c] : tmp[((size_t)(seq * PATCH + tok - 1) << 10) + c];
}

// ---------------- attention (RoPE + softmax + AV), tiny per head ----------------
// qkv: [NTOK, 1280] f32  (q:0..1023 | k:1024..1151 | v:1152..1279)
// ao : [NTOK, 1024] bf16
__global__ __launch_bounds__(256)
void attention(const float* __restrict__ qkv, unsigned short* __restrict__ ao)
{
  int t    = blockIdx.x * 256 + threadIdx.x;  // < NSEQ*NHEAD*QLEN
  int qpos = t % QLEN;
  int head = (t / QLEN) % NHEAD;
  int seq  = t / (QLEN * NHEAD);

  const float* qrow  = qkv + (size_t)(seq * QLEN + qpos) * 1280 + head * HD;
  const int kvh      = head >> 3;  // NH/KVH = 8
  const float* kbase = qkv + (size_t)(seq * QLEN) * 1280 + 1024 + kvh * HD;
  const float* vbase = kbase + 128;

  const float sc     = 0.35355339059327373f;   // HD^-0.25 baked into cos/sin
  const float neglog = -0.2878231366242557f;   // -ln(10000)/32

  float invf[32];
#pragma unroll
  for (int d = 0; d < 32; ++d) invf[d] = __expf(neglog * (float)d);

  // RoPE(q, pos=qpos)
  float q[64];
#pragma unroll
  for (int d = 0; d < 32; ++d) {
    float f = (float)qpos * invf[d];
    float s, c;
    __sincosf(f, &s, &c);
    c *= sc; s *= sc;
    float a = qrow[d], b = qrow[d + 32];
    q[d]      = a * c - b * s;
    q[d + 32] = b * c + a * s;
  }

  float logit[QLEN];
  for (int j = 0; j < QLEN; ++j) {
    const float* kr = kbase + (size_t)j * 1280;
    float accd = 0.0f;
#pragma unroll
    for (int d = 0; d < 32; ++d) {
      float f = (float)j * invf[d];
      float s, c;
      __sincosf(f, &s, &c);
      c *= sc; s *= sc;
      float a = kr[d], b = kr[d + 32];
      accd += q[d] * (a * c - b * s) + q[d + 32] * (b * c + a * s);
    }
    logit[j] = accd;
  }

  float mx = logit[0];
#pragma unroll
  for (int j = 1; j < QLEN; ++j) mx = fmaxf(mx, logit[j]);
  float p[QLEN], sum = 0.0f;
#pragma unroll
  for (int j = 0; j < QLEN; ++j) { p[j] = __expf(logit[j] - mx); sum += p[j]; }
  float rinv = 1.0f / sum;

  float o[64];
#pragma unroll
  for (int d = 0; d < 64; ++d) o[d] = 0.0f;
  for (int j = 0; j < QLEN; ++j) {
    float pj = p[j] * rinv;
    const float* vr = vbase + (size_t)j * 1280;
#pragma unroll
    for (int d = 0; d < 64; ++d) o[d] += pj * vr[d];
  }

  unsigned short* orow = ao + (size_t)(seq * QLEN + qpos) * HDIM + head * HD;
#pragma unroll
  for (int d = 0; d < 64; ++d) orow[d] = f2bf(o[d]);
}

// ---------------- host orchestration ----------------
extern "C" void kernel_launch(void* const* d_in, const int* in_sizes, int n_in,
                              void* d_out, int out_size, void* d_ws, size_t ws_size,
                              hipStream_t stream)
{
  (void)in_sizes; (void)n_in; (void)out_size; (void)ws_size;

  const float* audio   = (const float*)d_in[0];   // [8192,4,64]
  const float* special = (const float*)d_in[1];   // [1024]
  const float* in_proj = (const float*)d_in[2];   // [64,1024]
  const float* ln1     = (const float*)d_in[3];   // [4,1024]
  const float* wq      = (const float*)d_in[4];   // [4,1024,1024]
  const float* wk      = (const float*)d_in[5];   // [4,1024,128]
  const float* wv      = (const float*)d_in[6];   // [4,1024,128]
  const float* wo      = (const float*)d_in[7];   // [4,1024,1024]
  const float* ln2     = (const float*)d_in[8];   // [4,1024]
  const float* wg      = (const float*)d_in[9];   // [4,1024,4096]
  const float* wu      = (const float*)d_in[10];  // [4,1024,4096]
  const float* wd      = (const float*)d_in[11];  // [4,4096,1024]
  const float* fnw     = (const float*)d_in[12];  // [1024]
  const float* wlm     = (const float*)d_in[13];  // [1024,2048]
  float* out = (float*)d_out;                     // [1,8192,2048] f32

  char* ws = (char*)d_ws;
  size_t off = 0;
  auto alloc = [&](size_t bytes) -> char* {
    char* p = ws + off;
    off += (bytes + 255) & ~(size_t)255;
    return p;
  };

  unsigned short *qkvT[LAY], *woT[LAY], *gT[LAY], *uT[LAY], *dT[LAY];
  for (int l = 0; l < LAY; ++l) {
    qkvT[l] = (unsigned short*)alloc((size_t)1280 * 1024 * 2);
    woT[l]  = (unsigned short*)alloc((size_t)1024 * 1024 * 2);
    gT[l]   = (unsigned short*)alloc((size_t)4096 * 1024 * 2);
    uT[l]   = (unsigned short*)alloc((size_t)4096 * 1024 * 2);
    dT[l]   = (unsigned short*)alloc((size_t)1024 * 4096 * 2);
  }
  unsigned short* lmT     = (unsigned short*)alloc((size_t)2048 * 1024 * 2);
  unsigned short* ipT     = (unsigned short*)alloc((size_t)1024 * 64 * 2);
  unsigned short* audioBf = (unsigned short*)alloc((size_t)MROWS * 64 * 2);
  float*          h       = (float*)alloc((size_t)NTOK * HDIM * 4);
  unsigned short* xbf     = (unsigned short*)alloc((size_t)NTOK * HDIM * 2);
  float*          qkvbuf  = (float*)alloc((size_t)NTOK * 1280 * 4);   // also in_proj tmp
  unsigned short* aobf    = (unsigned short*)alloc((size_t)NTOK * HDIM * 2);
  unsigned short* mlpbf   = (unsigned short*)alloc((size_t)NTOK * FF * 2);
  unsigned short* x0bf    = (unsigned short*)alloc((size_t)NSEQ * HDIM * 2);

  const int T = 256;
  auto cdiv = [](long a, long b) { return (int)((a + b - 1) / b); };

  // ---- weight conversion (f32 [K][N] -> bf16 [N][K]) ----
  for (int l = 0; l < LAY; ++l) {
    tconv<<<cdiv(1024L * 1024, T), T, 0, stream>>>(wq + (size_t)l * 1024 * 1024, qkvT[l],                 1024, 1024, 1024 * 1024);
    tconv<<<cdiv(1024L * 128,  T), T, 0, stream>>>(wk + (size_t)l * 1024 * 128,  qkvT[l] + 1024 * 1024,   1024, 128,  1024 * 128);
    tconv<<<cdiv(1024L * 128,  T), T, 0, stream>>>(wv + (size_t)l * 1024 * 128,  qkvT[l] + 1152 * 1024,   1024, 128,  1024 * 128);
    tconv<<<cdiv(1024L * 1024, T), T, 0, stream>>>(wo + (size_t)l * 1024 * 1024, woT[l],                  1024, 1024, 1024 * 1024);
    tconv<<<cdiv(1024L * 4096, T), T, 0, stream>>>(wg + (size_t)l * 1024 * 4096, gT[l],                   1024, 4096, 1024 * 4096);
    tconv<<<cdiv(1024L * 4096, T), T, 0, stream>>>(wu + (size_t)l * 1024 * 4096, uT[l],                   1024, 4096, 1024 * 4096);
    tconv<<<cdiv(4096L * 1024, T), T, 0, stream>>>(wd + (size_t)l * 4096 * 1024, dT[l],                   4096, 1024, 4096 * 1024);
  }
  tconv<<<cdiv(1024L * 2048, T), T, 0, stream>>>(wlm,     lmT, 1024, 2048, 1024 * 2048);
  tconv<<<cdiv(64L * 1024,   T), T, 0, stream>>>(in_proj, ipT, 64,   1024, 64 * 1024);
  conv_bf16<<<cdiv((long)MROWS * 64, T), T, 0, stream>>>(audio, audioBf, MROWS * 64);

  // ---- in_proj GEMM + assemble h ----
  gemm_wmma<EPI_NONE><<<dim3(1024 / BTN, MROWS / BTM), 256, 0, stream>>>(
      audioBf, ipT, qkvbuf, nullptr, nullptr, MROWS, 1024, 64);
  build_h<<<cdiv((long)NTOK * HDIM, T), T, 0, stream>>>(qkvbuf, special, h);

  // ---- transformer layers ----
  for (int l = 0; l < LAY; ++l) {
    rms_bf16<<<NTOK, 256, 0, stream>>>(h, HDIM, ln1 + (size_t)l * HDIM, xbf);
    gemm_wmma<EPI_NONE><<<dim3(1280 / BTN, NTOK / BTM), 256, 0, stream>>>(
        xbf, qkvT[l], qkvbuf, nullptr, nullptr, NTOK, 1280, 1024);
    attention<<<cdiv((long)NSEQ * NHEAD * QLEN, T), T, 0, stream>>>(qkvbuf, aobf);
    gemm_wmma<EPI_RES><<<dim3(1024 / BTN, NTOK / BTM), 256, 0, stream>>>(
        aobf, woT[l], h, h, nullptr, NTOK, 1024, 1024);
    rms_bf16<<<NTOK, 256, 0, stream>>>(h, HDIM, ln2 + (size_t)l * HDIM, xbf);
    gemm_wmma<EPI_SILU><<<dim3(FF / BTN, NTOK / BTM), 256, 0, stream>>>(
        xbf, gT[l], nullptr, nullptr, mlpbf, NTOK, FF, 1024);
    gemm_wmma<EPI_MUL><<<dim3(FF / BTN, NTOK / BTM), 256, 0, stream>>>(
        xbf, uT[l], nullptr, nullptr, mlpbf, NTOK, FF, 1024);
    gemm_wmma<EPI_RES><<<dim3(1024 / BTN, NTOK / BTM), 256, 0, stream>>>(
        mlpbf, dT[l], h, h, nullptr, NTOK, 1024, FF);
  }

  // ---- final norm on token 0 + LM projection ----
  rms_bf16<<<NSEQ, 256, 0, stream>>>(h, QLEN * HDIM, fnw, x0bf);
  gemm_wmma<EPI_NONE><<<dim3(DLM / BTN, NSEQ / BTM), 256, 0, stream>>>(
      x0bf, lmT, out, nullptr, nullptr, NSEQ, DLM, 1024);
}
